// SimplifiedMambaSSM_49117245997238
// MI455X (gfx1250) — matrix operand order repaired and checked by reading, not compiled, not run on hardware
//
#include <hip/hip_runtime.h>

typedef __attribute__((ext_vector_type(16))) _Float16 v16h;
typedef __attribute__((ext_vector_type(8)))  float    v8f;
typedef __attribute__((ext_vector_type(4)))  _Float16 v4h;
typedef __attribute__((ext_vector_type(2)))  _Float16 h2;

#define D_MODEL 1024
#define D_STATE 16
#define BATCH   8
#define SEQ     4096
#define M_TOTAL (BATCH * SEQ)   // 32768 rows

// ---------------------------------------------------------------------------
// CDNA5 async global->LDS copy (tracked by ASYNCcnt), GV addressing mode.
// ---------------------------------------------------------------------------
__device__ __forceinline__ void async_ld_b128(unsigned lds_addr, const void* gptr) {
    asm volatile("global_load_async_to_lds_b128 %0, %1, off"
                 :: "v"(lds_addr), "v"(gptr)
                 : "memory");
}

template <int N>
__device__ __forceinline__ void s_wait_asynccnt_imm() {
    asm volatile("s_wait_asynccnt %0" :: "n"(N) : "memory");
}

// A-fragment K index used by this code (contiguous k order, same as its B loads; 16x32 MxK, f16):
//   pair p (elements e=2p,2p+1), lane-half h: K = 16*h + 2*p
__device__ __forceinline__ int a_frag_k(int p, int half) {
    return (half << 4) + (p << 1);
}

// ---------------------------------------------------------------------------
// Kernel 0: f32 -> f16 convert (feeds the async-staged WMMA kernels)
// ---------------------------------------------------------------------------
__global__ __launch_bounds__(256)
void cvt_f32_f16_kernel(const float* __restrict__ in, _Float16* __restrict__ out)
{
    int i = blockIdx.x * 256 + threadIdx.x;
    float4 v = reinterpret_cast<const float4*>(in)[i];
    v4h o = {(_Float16)v.x, (_Float16)v.y, (_Float16)v.z, (_Float16)v.w};
    reinterpret_cast<v4h*>(out)[i] = o;
}

// ===========================================================================
// Kernel 1: gating GEMM + sigmoid gate
//   xg[m,n] = sigmoid( sum_k x[m,k]*Wg[n,k] + bg[n] ) * x[m,n]   (f16 out)
// 128x128 block tile (8 waves, each 32x64 = 2x4 WMMA tiles), BK=32,
// double-buffered async-to-LDS staging, last iteration peeled.
// ===========================================================================
#define BM 128
#define BN 128
#define BK 32
#define TS 40   // LDS tile row stride in halves (80B = 5 x b128 slots, 4 used)

// Stage a [128 x 32]-half tile (row stride D_MODEL halves) into LDS.
// 512 x 16B chunks, 256 threads -> 2 async b128 copies per thread.
__device__ __forceinline__ void stage_tile(const _Float16* __restrict__ g,
                                           int row0, int k0,
                                           unsigned lds_base, int tid)
{
#pragma unroll
    for (int i = 0; i < 2; i++) {
        int c    = i * 256 + tid;   // chunk 0..511
        int row  = c >> 2;          // 0..127
        int part = c & 3;           // 16B part within the 64B row payload
        async_ld_b128(lds_base + row * (TS * 2) + part * 16,
                      g + (size_t)(row0 + row) * D_MODEL + k0 + part * 8);
    }
}

__global__ __launch_bounds__(256)
void gate_gemm_kernel(const _Float16* __restrict__ x16,
                      const _Float16* __restrict__ Wg16,
                      const float* __restrict__ bg, _Float16* __restrict__ xg)
{
    __shared__ _Float16 sX[2][BM * TS];
    __shared__ _Float16 sW[2][BN * TS];

    const int tid  = threadIdx.x;
    const int lane = tid & 31;
    const int wave = tid >> 5;
    const int wm   = wave >> 1;   // 0..3
    const int wn   = wave & 1;    // 0..1
    const int half = lane >> 4;
    const int l16  = lane & 15;

    const int m0 = blockIdx.y * BM;
    const int n0 = blockIdx.x * BN;

    const unsigned sxb[2] = {(unsigned)(size_t)&sX[0][0], (unsigned)(size_t)&sX[1][0]};
    const unsigned swb[2] = {(unsigned)(size_t)&sW[0][0], (unsigned)(size_t)&sW[1][0]};

    const v8f vzero = {0.f,0.f,0.f,0.f,0.f,0.f,0.f,0.f};
    v8f acc[2][4];
#pragma unroll
    for (int i = 0; i < 2; i++)
#pragma unroll
        for (int j = 0; j < 4; j++) acc[i][j] = vzero;

    // gather fragments from LDS buffer `cur` and issue the 8 WMMAs
    auto compute = [&](int cur) {
        v16h aF[2], bF[4];
#pragma unroll
        for (int sm = 0; sm < 2; sm++) {
            const _Float16* base = &sX[cur][(wm * 32 + sm * 16 + l16) * TS];
#pragma unroll
            for (int p = 0; p < 8; p++) {
                h2 v = *reinterpret_cast<const h2*>(base + a_frag_k(p, half));
                aF[sm][2*p]   = v.x;
                aF[sm][2*p+1] = v.y;
            }
        }
#pragma unroll
        for (int sn = 0; sn < 4; sn++) {
            // B 32x16: lane holds column n=l16; K = 16*half + e
            const _Float16* base = &sW[cur][(wn * 64 + sn * 16 + l16) * TS] + half * 16;
#pragma unroll
            for (int p = 0; p < 8; p++) {
                h2 v = *reinterpret_cast<const h2*>(base + 2 * p);
                bF[sn][2*p]   = v.x;
                bF[sn][2*p+1] = v.y;
            }
        }
#pragma unroll
        for (int sm = 0; sm < 2; sm++)
#pragma unroll
            for (int sn = 0; sn < 4; sn++)
                acc[sm][sn] = __builtin_amdgcn_wmma_f32_16x16x32_f16(
                    false, aF[sm], false, bF[sn],
                    (short)0, acc[sm][sn], false, false);
    };

    // prologue: stage k0 = 0 into buffer 0
    stage_tile(x16,  m0, 0, sxb[0], tid);
    stage_tile(Wg16, n0, 0, swb[0], tid);

    int cur = 0;
    // steady state: unconditional stage-next / wait<=4 / compute
    for (int k0 = 0; k0 < D_MODEL - BK; k0 += BK) {
        stage_tile(x16,  m0, k0 + BK, sxb[cur ^ 1], tid);
        stage_tile(Wg16, n0, k0 + BK, swb[cur ^ 1], tid);
        s_wait_asynccnt_imm<4>();   // in-order completion: current buffer ready
        __syncthreads();
        compute(cur);
        __syncthreads();            // done reading `cur` before it is restaged
        cur ^= 1;
    }
    // drain: last tile
    s_wait_asynccnt_imm<0>();
    __syncthreads();
    compute(cur);

    // ---- epilogue: sigmoid gate * x (f16 source), v_rcp instead of IEEE div
    // C/D layout: VGPR r -> M = r + 8*half, N = l16
#pragma unroll
    for (int sm = 0; sm < 2; sm++) {
#pragma unroll
        for (int sn = 0; sn < 4; sn++) {
            int n = n0 + wn * 64 + sn * 16 + l16;
            float bgn = bg[n];
#pragma unroll
            for (int r = 0; r < 8; r++) {
                int m = m0 + wm * 32 + sm * 16 + r + half * 8;
                float z  = acc[sm][sn][r] + bgn;
                float g  = __builtin_amdgcn_rcpf(1.0f + __expf(-z));
                float xv = (float)x16[(size_t)m * D_MODEL + n];
                xg[(size_t)m * D_MODEL + n] = (_Float16)(g * xv);
            }
        }
    }
}

// ===========================================================================
// Kernel 2: Bt = xg @ Wb^T, Ct = xg @ Wc^T   ([M,16] each, f32 accum)
// Block = 128 rows (8 waves x 16 rows); Wb/Wc LDS-resident as f16 [k][n];
// xg tiles double-buffer async-staged, last iteration peeled.
// ===========================================================================
__global__ __launch_bounds__(256)
void proj_bc_kernel(const _Float16* __restrict__ xg,
                    const float* __restrict__ Wb, const float* __restrict__ Wc,
                    float* __restrict__ Bt, float* __restrict__ Ct)
{
    __shared__ _Float16 sWb[D_MODEL * D_STATE]; // [k][n]
    __shared__ _Float16 sWc[D_MODEL * D_STATE];
    __shared__ _Float16 sXg[2][128 * TS];

    const int tid = threadIdx.x;
    const unsigned sxb[2] = {(unsigned)(size_t)&sXg[0][0], (unsigned)(size_t)&sXg[1][0]};
    const int m0b = blockIdx.x * 128;

    // preload weights (coalesced f32 read, transposed f16 store)
#pragma unroll
    for (int j = 0; j < 64; j++) {
        int f = j * 256 + tid;      // flat = n*1024 + k
        int n = f >> 10, k = f & 1023;
        sWb[k * D_STATE + n] = (_Float16)Wb[f];
        sWc[k * D_STATE + n] = (_Float16)Wc[f];
    }

    stage_tile(xg, m0b, 0, sxb[0], tid);   // 2 async b128 per thread

    const int lane = tid & 31, wave = tid >> 5;
    const int half = lane >> 4, l16 = lane & 15;
    const int mrow = wave * 16 + l16;      // row within block tile

    const v8f vzero = {0.f,0.f,0.f,0.f,0.f,0.f,0.f,0.f};
    v8f accB = vzero, accC = vzero;

    auto compute = [&](int cur, int k0) {
        v16h aF, bB, bC;
        const _Float16* base = &sXg[cur][mrow * TS];
#pragma unroll
        for (int p = 0; p < 8; p++) {
            h2 v = *reinterpret_cast<const h2*>(base + a_frag_k(p, half));
            aF[2*p]   = v.x;
            aF[2*p+1] = v.y;
        }
        int kb = k0 + half * 16;
#pragma unroll
        for (int e = 0; e < 16; e++) {
            bB[e] = sWb[(kb + e) * D_STATE + l16];
            bC[e] = sWc[(kb + e) * D_STATE + l16];
        }
        accB = __builtin_amdgcn_wmma_f32_16x16x32_f16(false, aF, false, bB,
                                                      (short)0, accB, false, false);
        accC = __builtin_amdgcn_wmma_f32_16x16x32_f16(false, aF, false, bC,
                                                      (short)0, accC, false, false);
    };

    int cur = 0;
    for (int k0 = 0; k0 < D_MODEL - 32; k0 += 32) {
        stage_tile(xg, m0b, k0 + 32, sxb[cur ^ 1], tid);
        s_wait_asynccnt_imm<2>();
        __syncthreads();
        compute(cur, k0);
        __syncthreads();
        cur ^= 1;
    }
    s_wait_asynccnt_imm<0>();
    __syncthreads();
    compute(cur, D_MODEL - 32);

#pragma unroll
    for (int r = 0; r < 8; r++) {
        int m = m0b + wave * 16 + r + half * 8;
        Bt[(size_t)m * D_STATE + l16] = accB[r];
        Ct[(size_t)m * D_STATE + l16] = accC[r];
    }
}

// ===========================================================================
// Kernel 3: sequential scan. One wave per batch; lanes carry h[n] (n=lane&15,
// upper half mirrors lower so xor-reductions stay uniform).
// ===========================================================================
__global__ __launch_bounds__(32)
void scan_kernel(const float* __restrict__ Bt, const float* __restrict__ Ct,
                 const float* __restrict__ A,  const float* __restrict__ h0,
                 float* __restrict__ ys, float* __restrict__ hfin)
{
    const int bat  = blockIdx.x;
    const int lane = threadIdx.x;
    const int n    = lane & 15;

    float s = 0.f;
    for (int d = 0; d < D_MODEL; d++) s += A[d * D_STATE + n];
    const float amean = s * (1.0f / (float)D_MODEL);

    float h = h0[bat * D_STATE + n];
    const float* Bb = Bt + (size_t)bat * SEQ * D_STATE;
    const float* Cb = Ct + (size_t)bat * SEQ * D_STATE;
    float* yb = ys + (size_t)bat * SEQ;

    for (int t = 0; t < SEQ; t++) {
        float b = Bb[t * D_STATE + n];
        float c = Cb[t * D_STATE + n];
        h = tanhf(h * amean + b);
        float p = c * h;
        p += __shfl_xor(p, 8, 32);
        p += __shfl_xor(p, 4, 32);
        p += __shfl_xor(p, 2, 32);
        p += __shfl_xor(p, 1, 32);
        if (lane == 0) yb[t] = p;
    }
    if (lane < 16) hfin[bat * D_STATE + n] = h;
}

// ===========================================================================
// Kernel 4: out[b,s,d] = ys[b,s] + D[d]*x[b,s,d]   (float4 vectorized, f32-exact)
// ===========================================================================
__global__ __launch_bounds__(256)
void output_kernel(const float* __restrict__ x, const float* __restrict__ Dv,
                   const float* __restrict__ ys, float* __restrict__ out)
{
    int i4 = blockIdx.x * 256 + threadIdx.x;
    const float4* x4 = reinterpret_cast<const float4*>(x);
    const float4* D4 = reinterpret_cast<const float4*>(Dv);
    float4*       o4 = reinterpret_cast<float4*>(out);
    int row  = i4 >> 8;
    int dcol = i4 & 255;
    float  y  = ys[row];
    float4 xv = x4[i4];
    float4 dv = D4[dcol];
    float4 ov;
    ov.x = y + dv.x * xv.x;
    ov.y = y + dv.y * xv.y;
    ov.z = y + dv.z * xv.z;
    ov.w = y + dv.w * xv.w;
    o4[i4] = ov;
}

// ===========================================================================
extern "C" void kernel_launch(void* const* d_in, const int* in_sizes, int n_in,
                              void* d_out, int out_size, void* d_ws, size_t ws_size,
                              hipStream_t stream)
{
    (void)in_sizes; (void)n_in; (void)out_size; (void)ws_size;

    const float* x  = (const float*)d_in[0];
    const float* h0 = (const float*)d_in[1];
    const float* A  = (const float*)d_in[2];
    const float* Wb = (const float*)d_in[3];
    const float* Wc = (const float*)d_in[4];
    const float* Dv = (const float*)d_in[5];
    const float* Wg = (const float*)d_in[6];
    const float* bg = (const float*)d_in[7];

    float* out = (float*)d_out;

    // workspace layout
    char* ws = (char*)d_ws;
    size_t off = 0;
    _Float16* xg   = (_Float16*)(ws + off); off += (size_t)M_TOTAL * D_MODEL * 2;  // 64 MiB
    _Float16* x16  = (_Float16*)(ws + off); off += (size_t)M_TOTAL * D_MODEL * 2;  // 64 MiB
    _Float16* Wg16 = (_Float16*)(ws + off); off += (size_t)D_MODEL * D_MODEL * 2;  //  2 MiB
    float* Bt = (float*)(ws + off); off += (size_t)M_TOTAL * D_STATE * 4;          //  2 MiB
    float* Ct = (float*)(ws + off); off += (size_t)M_TOTAL * D_STATE * 4;          //  2 MiB
    float* ys = (float*)(ws + off);
    float* hfin = out + (size_t)M_TOTAL * D_MODEL;   // tail of d_out (tuple order)

    cvt_f32_f16_kernel<<<(M_TOTAL * D_MODEL / 4) / 256, 256, 0, stream>>>(x, x16);
    cvt_f32_f16_kernel<<<(D_MODEL * D_MODEL / 4) / 256, 256, 0, stream>>>(Wg, Wg16);
    gate_gemm_kernel<<<dim3(D_MODEL / BN, M_TOTAL / BM), 256, 0, stream>>>(x16, Wg16, bg, xg);
    proj_bc_kernel<<<M_TOTAL / 128, 256, 0, stream>>>(xg, Wb, Wc, Bt, Ct);
    scan_kernel<<<BATCH, 32, 0, stream>>>(Bt, Ct, A, h0, ys, hfin);
    output_kernel<<<(M_TOTAL * D_MODEL / 4) / 256, 256, 0, stream>>>(x, Dv, ys, out);
}
